// MultiHeadAttention_1322849927297
// MI455X (gfx1250) — matrix operand order repaired and checked by reading, not compiled, not run on hardware
//
#include <hip/hip_runtime.h>
#include <hip/hip_bf16.h>

#define DMODEL 1024
#define SLEN   2048
#define NB     2
#define NH     16
#define DHEAD  64
#define MTOT   (NB*SLEN)   // 4096

typedef __attribute__((ext_vector_type(16))) _Float16     v16h;
typedef __attribute__((ext_vector_type(8)))  float        v8f;
typedef __attribute__((ext_vector_type(4)))  unsigned int v4u;

union Frag { v16h h; v4u u[2]; };

__device__ __forceinline__ v8f wmma16(v16h a, v16h b, v8f c) {
  // D = A(16x32 f16) x B(32x16 f16) + C(16x16 f32)
  return __builtin_amdgcn_wmma_f32_16x16x32_f16(false, a, false, b, (short)0, c,
                                                false, false);
}

__device__ __forceinline__ v8f v8f_zero() {
  v8f z;
#pragma unroll
  for (int i = 0; i < 8; ++i) z[i] = 0.0f;
  return z;
}

// A-fragment: 16 contiguous halves at p (16*half baked into p by caller)
__device__ __forceinline__ void ldA(Frag& f, const _Float16* p) {
  f.u[0] = *(const v4u*)(p);
  f.u[1] = *(const v4u*)(p + 8);
}
// B-fragment: 16 contiguous halves at p (16*half baked into p by caller)
__device__ __forceinline__ void ldB(Frag& f, const _Float16* p) {
  f.u[0] = *(const v4u*)(p);
  f.u[1] = *(const v4u*)(p + 8);
}

// ---------------------------------------------------------------------------
// f32 -> f16 conversion (x)
// ---------------------------------------------------------------------------
__global__ void cvt_f32_f16(const float* __restrict__ src,
                            _Float16* __restrict__ dst, int n) {
  int i = (blockIdx.x * blockDim.x + threadIdx.x) * 4;
  if (i + 3 < n) {
    float4 v = *(const float4*)(src + i);
    dst[i + 0] = (_Float16)v.x;
    dst[i + 1] = (_Float16)v.y;
    dst[i + 2] = (_Float16)v.z;
    dst[i + 3] = (_Float16)v.w;
  }
}

// ---------------------------------------------------------------------------
// Weight transpose + convert: Wt[n][k] = (f16) W[k][n]   (1024x1024)
// ---------------------------------------------------------------------------
__global__ void transposeW(const float* __restrict__ W,
                           _Float16* __restrict__ Wt) {
  __shared__ float tile[16][17];
  const int tx = threadIdx.x & 15;
  const int ty = threadIdx.x >> 4;
  const int bx = blockIdx.x;   // n / 16
  const int by = blockIdx.y;   // k / 16
  tile[ty][tx] = W[(by * 16 + ty) * DMODEL + bx * 16 + tx];
  __syncthreads();
  Wt[(bx * 16 + ty) * DMODEL + by * 16 + tx] = (_Float16)tile[tx][ty];
}

// ---------------------------------------------------------------------------
// WMMA GEMM:  out[m,n] = A[m,:] . Bt[n,:] + bias[n]      (M=4096, N=K=1024)
// One wave produces a 32x64 tile (2x4 accumulators), K-loop of 32, with
// register double-buffering so fragment loads overlap the WMMAs.
// mode 0: f16 out at [B,H,S,Dh]   (Q, K)
// mode 1: f16 out at [B,H,Dh,S]   (V transposed)
// mode 2: f32 out row-major [M,N] (final projection)
// ---------------------------------------------------------------------------
__global__ __launch_bounds__(128) void gemm_wmma(
    const _Float16* __restrict__ A,    // [MTOT, DMODEL]
    const _Float16* __restrict__ Bt,   // [DMODEL(n), DMODEL(k)]
    const float* __restrict__ bias,    // [DMODEL]
    _Float16* __restrict__ outH, float* __restrict__ outF, int mode) {
  const int lane = threadIdx.x & 31;
  const int wave = threadIdx.x >> 5;
  const int tile = blockIdx.x * 4 + wave;
  const int NT64 = DMODEL / 64;                 // 16 n-tiles
  const int mbase = (tile / NT64) * 32;
  const int nbase = (tile % NT64) * 64;
  const int half = lane >> 4;
  const int lq = lane & 15;

  const _Float16* arow0 = A + (size_t)(mbase + lq) * DMODEL + 16 * half;
  const _Float16* arow1 = arow0 + (size_t)16 * DMODEL;
  const _Float16* brow0 = Bt + (size_t)(nbase + 0  + lq) * DMODEL + 16 * half;
  const _Float16* brow1 = Bt + (size_t)(nbase + 16 + lq) * DMODEL + 16 * half;
  const _Float16* brow2 = Bt + (size_t)(nbase + 32 + lq) * DMODEL + 16 * half;
  const _Float16* brow3 = Bt + (size_t)(nbase + 48 + lq) * DMODEL + 16 * half;

  v8f acc[2][4];
#pragma unroll
  for (int mt = 0; mt < 2; ++mt)
#pragma unroll
    for (int t = 0; t < 4; ++t) acc[mt][t] = v8f_zero();

  // prologue: fragments for kb = 0
  Frag a0, a1, b0, b1, b2, b3;
  ldA(a0, arow0); ldA(a1, arow1);
  ldB(b0, brow0); ldB(b1, brow1); ldB(b2, brow2); ldB(b3, brow3);

  for (int kb = 0; kb < DMODEL; kb += 32) {
    const int kn = kb + 32;
    Frag na0, na1, nb0, nb1, nb2, nb3;
    if (kn < DMODEL) {   // issue next iteration's loads BEFORE the WMMAs
      ldA(na0, arow0 + kn); ldA(na1, arow1 + kn);
      ldB(nb0, brow0 + kn); ldB(nb1, brow1 + kn);
      ldB(nb2, brow2 + kn); ldB(nb3, brow3 + kn);
      __builtin_prefetch(arow0 + kn + 256, 0, 0);   // global_prefetch_b8
    }
    acc[0][0] = wmma16(a0.h, b0.h, acc[0][0]);
    acc[1][0] = wmma16(a1.h, b0.h, acc[1][0]);
    acc[0][1] = wmma16(a0.h, b1.h, acc[0][1]);
    acc[1][1] = wmma16(a1.h, b1.h, acc[1][1]);
    acc[0][2] = wmma16(a0.h, b2.h, acc[0][2]);
    acc[1][2] = wmma16(a1.h, b2.h, acc[1][2]);
    acc[0][3] = wmma16(a0.h, b3.h, acc[0][3]);
    acc[1][3] = wmma16(a1.h, b3.h, acc[1][3]);
    a0 = na0; a1 = na1;
    b0 = nb0; b1 = nb1; b2 = nb2; b3 = nb3;
  }

#pragma unroll
  for (int mt = 0; mt < 2; ++mt) {
#pragma unroll
    for (int t = 0; t < 4; ++t) {
#pragma unroll
      for (int r = 0; r < 8; ++r) {
        const int m = mbase + mt * 16 + r + 8 * half;  // C layout row
        const int n = nbase + t * 16 + lq;             // C layout col
        const float v = acc[mt][t][r] + bias[n];
        if (mode == 0) {
          const int b = m >> 11, s = m & (SLEN - 1);
          const int h = n >> 6, d = n & 63;
          outH[(((size_t)(b * NH + h) * SLEN + s) << 6) + d] = (_Float16)v;
        } else if (mode == 1) {
          const int b = m >> 11, s = m & (SLEN - 1);
          const int h = n >> 6, d = n & 63;
          outH[((size_t)(b * NH + h) * DHEAD + d) * SLEN + s] = (_Float16)v;
        } else {
          outF[(size_t)m * DMODEL + n] = v;
        }
      }
    }
  }
}

// ---------------------------------------------------------------------------
// Flash attention: one wave per (b,h, 16-query tile). Causal, fp32 softmax.
// K fragments double-buffered across kb; V loads hoisted before softmax so
// the exp/shuffle VALU work hides their latency.
// ---------------------------------------------------------------------------
__global__ __launch_bounds__(128) void attn_wmma(
    const _Float16* __restrict__ Q,    // [B*H, S, 64]
    const _Float16* __restrict__ K,    // [B*H, S, 64]
    const _Float16* __restrict__ Vt,   // [B*H, 64, S]
    _Float16* __restrict__ ctx) {      // [B*S, 1024]
  __shared__ __align__(16) _Float16 pb[4][16 * 32];

  const int lane = threadIdx.x & 31;
  const int wave = threadIdx.x >> 5;
  const int tile = blockIdx.x * 4 + wave;
  const int qtile = tile & 127;  // S/16 = 128 tiles
  const int bh = tile >> 7;
  const int qbase = qtile * 16;
  const int half = lane >> 4;
  const int lq = lane & 15;

  const _Float16* Qh = Q + (size_t)bh * SLEN * DHEAD;
  const _Float16* Kh = K + (size_t)bh * SLEN * DHEAD;
  const _Float16* Vh = Vt + (size_t)bh * DHEAD * SLEN;
  _Float16* pbw = &pb[wave][0];

  // Q tile as two 16x32 A-fragments (d = 0..31, 32..63)
  Frag qa0, qa1;
  {
    const _Float16* qr = Qh + (size_t)(qbase + lq) * DHEAD + 16 * half;
    ldA(qa0, qr);
    ldA(qa1, qr + 32);
  }

  v8f acc[4];
#pragma unroll
  for (int t = 0; t < 4; ++t) acc[t] = v8f_zero();
  float mrow[8], lrow[8];
#pragma unroll
  for (int r = 0; r < 8; ++r) {
    mrow[r] = -__builtin_inff();
    lrow[r] = 0.0f;
  }

  const int kend = qbase + 16;

  // prologue: K fragments for kb = 0
  Frag k00, k01, k10, k11;
  {
    const _Float16* kp0 = Kh + (size_t)lq * DHEAD + 16 * half;
    const _Float16* kp1 = Kh + (size_t)(16 + lq) * DHEAD + 16 * half;
    ldB(k00, kp0); ldB(k01, kp0 + 32);
    ldB(k10, kp1); ldB(k11, kp1 + 32);
  }

  for (int kb = 0; kb < kend; kb += 32) {  // 32 keys per step (causal)
    // ---- scores: two 16x16 tiles (keys kb..kb+15, kb+16..kb+31) ----
    v8f s0 = v8f_zero(), s1 = v8f_zero();
    s0 = wmma16(qa0.h, k00.h, s0);
    s1 = wmma16(qa0.h, k10.h, s1);
    s0 = wmma16(qa1.h, k01.h, s0);
    s1 = wmma16(qa1.h, k11.h, s1);

    // ---- double-buffer: next iteration's K fragments ----
    const int kn = kb + 32;
    Frag nk00, nk01, nk10, nk11;
    if (kn < kend) {
      const _Float16* kp0 = Kh + (size_t)(kn + lq) * DHEAD + 16 * half;
      const _Float16* kp1 = Kh + (size_t)(kn + 16 + lq) * DHEAD + 16 * half;
      ldB(nk00, kp0); ldB(nk01, kp0 + 32);
      ldB(nk10, kp1); ldB(nk11, kp1 + 32);
    }

    // ---- V fragments for THIS kb, issued before softmax to overlap ----
    Frag vb0, vb1, vb2, vb3;
    {
      const _Float16* vp = Vh + (size_t)lq * SLEN + kb + 16 * half;
      ldB(vb0, vp);
      ldB(vb1, vp + (size_t)16 * SLEN);
      ldB(vb2, vp + (size_t)32 * SLEN);
      ldB(vb3, vp + (size_t)48 * SLEN);
    }

    // ---- mask, scale, online softmax ----
#pragma unroll
    for (int r = 0; r < 8; ++r) {
      const int m = qbase + r + 8 * half;
      float v0 = s0[r], v1 = s1[r];
      if (kb + lq > m) v0 = -1e9f;        // mask BEFORE scaling (ref order)
      if (kb + 16 + lq > m) v1 = -1e9f;
      v0 *= 0.125f;                       // 1/sqrt(64)
      v1 *= 0.125f;
      float mx = fmaxf(v0, v1);
#pragma unroll
      for (int o = 1; o < 16; o <<= 1)
        mx = fmaxf(mx, __shfl_xor(mx, o, 32));  // row max across 16 lanes
      const float mnew = fmaxf(mrow[r], mx);
      const float alpha = __expf(mrow[r] - mnew);
      const float p0 = __expf(v0 - mnew);
      const float p1 = __expf(v1 - mnew);
      float rs = p0 + p1;
#pragma unroll
      for (int o = 1; o < 16; o <<= 1) rs += __shfl_xor(rs, o, 32);
      lrow[r] = lrow[r] * alpha + rs;
      mrow[r] = mnew;
#pragma unroll
      for (int t = 0; t < 4; ++t) acc[t][r] *= alpha;
      // stage P (f16) into LDS, row-major [16 rows][32 keys]
      const int mb = (r + 8 * half) * 32;
      pbw[mb + lq] = (_Float16)p0;
      pbw[mb + 16 + lq] = (_Float16)p1;
    }

    // ---- re-layout P from LDS into a 16x32 A-fragment ----
    Frag pa;
    pa.u[0] = *(const v4u*)&pbw[lq * 32 + 16 * half];
    pa.u[1] = *(const v4u*)&pbw[lq * 32 + 8 + 16 * half];

    // ---- ctx += P (16x32) . V (32x64) ----
    acc[0] = wmma16(pa.h, vb0.h, acc[0]);
    acc[1] = wmma16(pa.h, vb1.h, acc[1]);
    acc[2] = wmma16(pa.h, vb2.h, acc[2]);
    acc[3] = wmma16(pa.h, vb3.h, acc[3]);

    k00 = nk00; k01 = nk01; k10 = nk10; k11 = nk11;
  }

  // ---- normalize and write ctx as f16 [B*S, 1024] ----
  const int b = bh >> 4, h = bh & 15;
#pragma unroll
  for (int t = 0; t < 4; ++t) {
#pragma unroll
    for (int r = 0; r < 8; ++r) {
      const int m = qbase + r + 8 * half;
      const float v = acc[t][r] / lrow[r];
      ctx[(size_t)(b * SLEN + m) * DMODEL + h * 64 + t * 16 + lq] =
          (_Float16)v;
    }
  }
}

// ---------------------------------------------------------------------------
// Host launcher
// ---------------------------------------------------------------------------
extern "C" void kernel_launch(void* const* d_in, const int* in_sizes, int n_in,
                              void* d_out, int out_size, void* d_ws,
                              size_t ws_size, hipStream_t stream) {
  const float* x  = (const float*)d_in[0];
  const float* Wq = (const float*)d_in[1];
  const float* bq = (const float*)d_in[2];
  const float* Wk = (const float*)d_in[3];
  const float* bk = (const float*)d_in[4];
  const float* Wv = (const float*)d_in[5];
  const float* bv = (const float*)d_in[6];
  const float* Wo = (const float*)d_in[7];
  const float* bo = (const float*)d_in[8];
  float* out = (float*)d_out;

  char* ws = (char*)d_ws;
  const size_t MB = 1024 * 1024;
  _Float16* xh   = (_Float16*)(ws + 0 * MB);    // 8 MB  [4096,1024]
  _Float16* WqT  = (_Float16*)(ws + 8 * MB);    // 2 MB  [n][k]
  _Float16* WkT  = (_Float16*)(ws + 10 * MB);
  _Float16* WvT  = (_Float16*)(ws + 12 * MB);
  _Float16* WoT  = (_Float16*)(ws + 14 * MB);
  _Float16* Qh   = (_Float16*)(ws + 16 * MB);   // 8 MB  [B,H,S,Dh]
  _Float16* Kh   = (_Float16*)(ws + 24 * MB);   // 8 MB  [B,H,S,Dh]
  _Float16* Vt   = (_Float16*)(ws + 32 * MB);   // 8 MB  [B,H,Dh,S]
  _Float16* ctxh = (_Float16*)(ws + 40 * MB);   // 8 MB  [4096,1024]

  const int nX = MTOT * DMODEL;
  cvt_f32_f16<<<nX / (256 * 4), 256, 0, stream>>>(x, xh, nX);

  dim3 tg(DMODEL / 16, DMODEL / 16);
  transposeW<<<tg, 256, 0, stream>>>(Wq, WqT);
  transposeW<<<tg, 256, 0, stream>>>(Wk, WkT);
  transposeW<<<tg, 256, 0, stream>>>(Wv, WvT);
  transposeW<<<tg, 256, 0, stream>>>(Wo, WoT);

  const int gemmBlocks = (MTOT / 32) * (DMODEL / 64) / 4;  // 512
  gemm_wmma<<<gemmBlocks, 128, 0, stream>>>(xh, WqT, bq, Qh, nullptr, 0);
  gemm_wmma<<<gemmBlocks, 128, 0, stream>>>(xh, WkT, bk, Kh, nullptr, 0);
  gemm_wmma<<<gemmBlocks, 128, 0, stream>>>(xh, WvT, bv, Vt, nullptr, 1);

  const int attnBlocks = NB * NH * (SLEN / 16) / 4;        // 1024
  attn_wmma<<<attnBlocks, 128, 0, stream>>>(Qh, Kh, Vt, ctxh);

  gemm_wmma<<<gemmBlocks, 128, 0, stream>>>(ctxh, WoT, bo, nullptr, out, 2);
}
